// Generator_66554813219242
// MI455X (gfx1250) — matrix operand order repaired and checked by reading, not compiled, not run on hardware
//
#include <hip/hip_runtime.h>
#include <math.h>

// ============================ device helpers ============================

#define SQRT2F 1.41421356237309515f

typedef __attribute__((ext_vector_type(16))) __bf16  v16bf;
typedef __attribute__((ext_vector_type(8)))  float   v8f;
typedef __attribute__((ext_vector_type(4)))  unsigned int v4u;

union Frag32 { v4u q[2]; v16bf bf; };

__device__ __forceinline__ unsigned short f2bf(float f) {
  unsigned u = __float_as_uint(f);
  u += 0x7FFFu + ((u >> 16) & 1u);          // round-to-nearest-even
  return (unsigned short)(u >> 16);
}
__device__ __forceinline__ float bf2f(unsigned short h) {
  return __uint_as_float(((unsigned)h) << 16);
}
__device__ __forceinline__ float flrelu(float v) {
  return (v > 0.f ? v : 0.2f * v) * SQRT2F;
}

// ============================ mapping network ============================
// z[8,64] -> PixelNorm -> 8x EqualLinear(lr_mul=0.01)+flrelu -> latent[8,320]
struct P8 { const float* p[8]; };

__global__ void mapping_k(const float* __restrict__ label, const float* __restrict__ z,
                          P8 W, P8 Bb, float* __restrict__ latent) {
  __shared__ float zz[8][64];
  __shared__ float nrm[8];
  int tid = threadIdx.x;            // blockDim == 512
  int b = tid >> 6, j = tid & 63;
  zz[b][j] = z[b * 64 + j];
  __syncthreads();
  if (tid < 8) {
    float s = 0.f;
    for (int k = 0; k < 64; ++k) { float v = zz[tid][k]; s += v * v; }
    nrm[tid] = rsqrtf(s * (1.0f / 64.0f) + 1e-8f);
  }
  __syncthreads();
  float cur = zz[b][j] * nrm[b];
  __syncthreads();
  zz[b][j] = cur;
  __syncthreads();
  const float msc = (1.0f / 8.0f) * 0.01f;   // (1/sqrt(64)) * lr_mul
  for (int l = 0; l < 8; ++l) {
    const float* w = W.p[l];
    const float* bi = Bb.p[l];
    float s = 0.f;
    for (int k = 0; k < 64; ++k) s += zz[b][k] * w[j * 64 + k];
    s = flrelu(s * msc + bi[j] * 0.01f);
    __syncthreads();
    zz[b][j] = s;
    __syncthreads();
  }
  for (int idx = tid; idx < 8 * 320; idx += 512) {
    int bb = idx / 320, k = idx % 320;       // constant divisors
    latent[idx] = (k < 256) ? label[bb * 256 + k] : zz[bb][k - 256];
  }
}

// ============================ per-layer small kernels ============================
// All runtime strides are powers of two -> passed as log2 shifts (no software division).

// styleS[b,i] = (latent[b,:] . mw[i,:] / sqrt(320) + mb[i]) * conv_scale
__global__ void style_k(const float* __restrict__ latent, const float* __restrict__ mw,
                        const float* __restrict__ mb, float* __restrict__ styleS,
                        int Ish, float cscale) {
  unsigned idx = blockIdx.x * blockDim.x + threadIdx.x;
  unsigned I = 1u << Ish;
  if (idx >= (8u << Ish)) return;
  unsigned b = idx >> Ish, i = idx & (I - 1u);
  const float* lp = latent + b * 320u;
  const float* wp = mw + i * 320u;
  float s = 0.f;
  for (int k = 0; k < 320; ++k) s += lp[k] * wp[k];
  styleS[idx] = (s * 0.055901699437f + mb[i]) * cscale;   // 1/sqrt(320)
}

// wsq[o*I+i] = sum_t w[o,i,t]^2      (w layout [O][I][3][3])
__global__ void wsq_k(const float* __restrict__ w, float* __restrict__ wsq, unsigned OI) {
  unsigned idx = blockIdx.x * blockDim.x + threadIdx.x;
  if (idx >= OI) return;
  const float* p = w + (size_t)idx * 9u;
  float s = 0.f;
  for (int t = 0; t < 9; ++t) s += p[t] * p[t];
  wsq[idx] = s;
}

// demod[b,o] = rsqrt( sum_i styleS^2 * wsq + 1e-8 )  (scale already folded into styleS)
__global__ void demod_k(const float* __restrict__ styleS, const float* __restrict__ wsq,
                        float* __restrict__ demod, int Ish, int Osh) {
  unsigned idx = blockIdx.x * blockDim.x + threadIdx.x;
  unsigned O = 1u << Osh, I = 1u << Ish;
  if (idx >= (8u << Osh)) return;
  unsigned b = idx >> Osh, o = idx & (O - 1u);
  const float* st = styleS + (b << Ish);
  const float* wq = wsq + (o << Ish);
  float s = 0.f;
  for (unsigned i = 0; i < I; ++i) s += st[i] * st[i] * wq[i];
  demod[idx] = rsqrtf(s + 1e-8f);
}

// pack weights f32 [O][I][3][3] -> bf16 [O][9][I]; flip both spatial dims if flip!=0
__global__ void pack_k(const float* __restrict__ w, unsigned short* __restrict__ wp,
                       int Ish, int flip, unsigned total) {
  unsigned idx = blockIdx.x * blockDim.x + threadIdx.x;
  if (idx >= total) return;
  unsigned I = 1u << Ish;
  unsigned i = idx & (I - 1u);
  unsigned r = idx >> Ish;
  unsigned t = r % 9u;                       // constant divisor
  unsigned o = r / 9u;
  unsigned tt = flip ? (8u - t) : t;
  wp[idx] = f2bf(w[((size_t)(o << Ish) + i) * 9u + tt]);
}

// xm[b,y,x,i] = src * styleS[b,i].   CONST_SRC: src = f32 NCHW [I][H][H] (batch-broadcast)
template <bool CONST_SRC>
__global__ void modulate_k(const void* __restrict__ src, const float* __restrict__ styleS,
                           unsigned short* __restrict__ xm, int HHsh, int Ish) {
  unsigned idx = blockIdx.x * blockDim.x + threadIdx.x;
  unsigned total = 8u << (HHsh + Ish);
  if (idx >= total) return;
  unsigned I = 1u << Ish, HH = 1u << HHsh;
  unsigned i = idx & (I - 1u);
  unsigned pi = idx >> Ish;
  unsigned pix = pi & (HH - 1u);
  unsigned b = pi >> HHsh;
  float a;
  if (CONST_SRC) a = ((const float*)src)[(i << HHsh) + pix];
  else           a = bf2f(((const unsigned short*)src)[idx]);
  xm[idx] = f2bf(a * styleS[(b << Ish) + i]);
}

// ============================ WMMA implicit-GEMM conv ============================
// D[o, p] = sum_{t, i} wpack[o][t][i] * xm[b][yi(p,t)][xi(p,t)][i]
// Non-UP: 3x3 pad-1 conv, fused demod + noise + bias + flrelu -> bf16 NHWC act.
// UP: conv_transpose stride2 (zero-stuffed gather, kernel pre-flipped), fused demod,
//     raw bf16 NHWC tmp at (2H+1)^2; blur/noise/act applied by blur_k.
//
// Grid: x = ceil(Ho/32) column super-tiles (2 N-tiles per wave sharing one A fragment:
// halves weight traffic and provides two independent accumulator chains so the XDL
// pipe is never stalled on a WMMA->WMMA RAW), y = Ho rows, z = 8 * (O/128).
// Block = 8 waves; wave w owns o-tile (og*8 + w)*16 -> zero integer division.
// Row-tap validity and the second-tile presence are wave-uniform, so every executed
// v_wmma sees EXEC == all-ones as the ISA requires.

template <bool UP>
__device__ __forceinline__ void conv_epilogue(
    v8f acc, int b, int y, int x, int Ho, int O, int o0, int hi,
    const float* __restrict__ demod, const float* __restrict__ noise,
    const float* __restrict__ ab, float nw, unsigned short* __restrict__ dst) {
  float nn = 0.f;
  if (!UP) nn = nw * noise[y * Ho + x];
  unsigned short hv[8];
#pragma unroll
  for (int r = 0; r < 8; ++r) {
    int o = o0 + hi * 8 + r;                  // C/D layout: VGPR r -> M = r + 8*hi
    float val = acc[r] * demod[(unsigned)b * (unsigned)O + (unsigned)o];
    if (!UP) val = flrelu(val + nn + ab[o]);
    hv[r] = f2bf(val);
  }
  v4u st;
  st[0] = (unsigned)hv[0] | ((unsigned)hv[1] << 16);
  st[1] = (unsigned)hv[2] | ((unsigned)hv[3] << 16);
  st[2] = (unsigned)hv[4] | ((unsigned)hv[5] << 16);
  st[3] = (unsigned)hv[6] | ((unsigned)hv[7] << 16);
  *(v4u*)(dst + ((unsigned)(b * Ho + y) * (unsigned)Ho + (unsigned)x) * (unsigned)O +
          (unsigned)(o0 + hi * 8)) = st;
}

template <bool UP>
__global__ __launch_bounds__(256) void conv_wmma_k(
    const unsigned short* __restrict__ xm,   // [8][H][H][I] bf16 (modulated)
    const unsigned short* __restrict__ wp,   // [O][9][I]    bf16
    const float* __restrict__ demod,         // [8][O]
    const float* __restrict__ noise,         // [Ho][Ho]  (non-UP only)
    const float* __restrict__ ab,            // [O]       (non-UP only)
    const float* __restrict__ nwp,           // scalar    (non-UP only)
    unsigned short* __restrict__ dst,        // [8][Ho][Ho][O] bf16
    int H, int I, int O, int otg_sh) {
  const int lane = threadIdx.x & 31;
  const int wav  = threadIdx.x >> 5;
  const int Ho = UP ? (2 * H + 1) : H;
  const unsigned z = blockIdx.z;
  const int b  = (int)(z >> otg_sh);
  const int og = (int)(z & ((1u << otg_sh) - 1u));
  const int o0 = (og * 8 + wav) << 4;
  const int y  = blockIdx.y;
  const int x0 = blockIdx.x << 5;             // 32 columns per block (2 N-tiles)

  const int hi = lane >> 4;                   // lane half selects K sub-range (ISA A/B layout)
  const int ln = lane & 15;
  const int xa = x0 + ln;
  const int xb = xa + 16;
  const bool haveB = (x0 + 16) < Ho;          // wave-uniform
  const bool pva = xa < Ho;
  const bool pvb = xb < Ho;

  v8f acc0 = {0.f, 0.f, 0.f, 0.f, 0.f, 0.f, 0.f, 0.f};
  v8f acc1 = {0.f, 0.f, 0.f, 0.f, 0.f, 0.f, 0.f, 0.f};
  const int Zs = 2 * H - 1;                   // zero-stuffed extent (UP path)

  for (int t = 0; t < 9; ++t) {
    const int ky = t / 3, kx = t - 3 * ky;    // constant
    int yi;
    bool rowv;                                // wave-uniform
    int xia, xib;
    bool cva, cvb;                            // per-lane column validity
    if (UP) {
      int u = y + ky - 2;
      rowv = (u >= 0) && (u < Zs) && !(u & 1);
      yi = u >> 1;
      int va = xa + kx - 2, vb = xb + kx - 2;
      cva = pva && (va >= 0) && (va < Zs) && !(va & 1);
      cvb = pvb && (vb >= 0) && (vb < Zs) && !(vb & 1);
      xia = va >> 1; xib = vb >> 1;
    } else {
      yi = y + ky - 1;
      rowv = (yi >= 0) && (yi < H);
      xia = xa + kx - 1; xib = xb + kx - 1;
      cva = pva && (xia >= 0) && (xia < H);
      cvb = pvb && (xib >= 0) && (xib < H);
    }
    if (!rowv) continue;                      // uniform skip: EXEC stays all-ones
    const unsigned rowb = (unsigned)(b * H + yi) * (unsigned)H;
    // B sources: 16 contiguous bf16 channels per spatial position (NHWC)
    const unsigned short* bba = xm + (rowb + (unsigned)(cva ? xia : 0)) * (unsigned)I +
                                (unsigned)(hi * 16);
    const unsigned short* bbb = xm + (rowb + (unsigned)(cvb ? xib : 0)) * (unsigned)I +
                                (unsigned)(hi * 16);
    // A source: contiguous K (matches B) — lane<16: K 0..15 ; lane>=16: K 16..31
    const unsigned short* abase = wp + ((unsigned)(o0 + ln) * 9u + (unsigned)t) * (unsigned)I +
                                  (unsigned)(hi * 16);
    for (int ic0 = 0; ic0 < I; ic0 += 32) {
      Frag32 A;
      A.q[0] = *(const v4u*)(abase + ic0);
      A.q[1] = *(const v4u*)(abase + ic0 + 8);
      Frag32 B0;
      if (cva) {
        B0.q[0] = *(const v4u*)(bba + ic0);
        B0.q[1] = *(const v4u*)(bba + ic0 + 8);
        __builtin_prefetch(bba + ic0 + 64, 0, 1);     // global_prefetch_b8: K block two ahead
      } else {
        B0.q[0] = (v4u)0u;
        B0.q[1] = (v4u)0u;
      }
      acc0 = __builtin_amdgcn_wmma_f32_16x16x32_bf16(
          false, A.bf, false, B0.bf, (short)0, acc0, false, false);
      if (haveB) {                                    // wave-uniform
        Frag32 B1;
        if (cvb) {
          B1.q[0] = *(const v4u*)(bbb + ic0);
          B1.q[1] = *(const v4u*)(bbb + ic0 + 8);
        } else {
          B1.q[0] = (v4u)0u;
          B1.q[1] = (v4u)0u;
        }
        acc1 = __builtin_amdgcn_wmma_f32_16x16x32_bf16(
            false, A.bf, false, B1.bf, (short)0, acc1, false, false);
      }
    }
  }

  float nw = 0.f;
  if (!UP) nw = nwp[0];
  if (pva)
    conv_epilogue<UP>(acc0, b, y, xa, Ho, O, o0, hi, demod, noise, ab, nw, dst);
  if (haveB && pvb)
    conv_epilogue<UP>(acc1, b, y, xb, Ho, O, o0, hi, demod, noise, ab, nw, dst);
}

// 4-tap FIR blur (pad 1) on conv-transpose output + noise + bias + flrelu -> bf16 act
// Hs = 2H+1 (odd source extent), Ho = 2H (pow2), O pow2.
__global__ void blur_k(const unsigned short* __restrict__ tmp, const float* __restrict__ noise,
                       const float* __restrict__ ab, const float* __restrict__ nwp,
                       unsigned short* __restrict__ act, int Hs, int Hosh, int Osh) {
  unsigned idx = blockIdx.x * blockDim.x + threadIdx.x;
  unsigned Ho = 1u << Hosh, O = 1u << Osh;
  unsigned total = 8u << (Osh + 2 * Hosh);
  if (idx >= total) return;
  unsigned o = idx & (O - 1u);
  unsigned pi = idx >> Osh;
  int x = (int)(pi & (Ho - 1u));
  int y = (int)((pi >> Hosh) & (Ho - 1u));
  unsigned b = pi >> (2 * Hosh);
  const float BB[4] = {1.f, 3.f, 3.f, 1.f};
  float s = 0.f;
  for (int ky = 0; ky < 4; ++ky) {
    int yy = y + ky - 1;
    if (yy < 0 || yy >= Hs) continue;
    for (int kx = 0; kx < 4; ++kx) {
      int xx = x + kx - 1;
      if (xx < 0 || xx >= Hs) continue;
      s += BB[ky] * BB[kx] *
           bf2f(tmp[((b * (unsigned)Hs + (unsigned)yy) * (unsigned)Hs + (unsigned)xx) * O + o]);
    }
  }
  s *= (1.0f / 16.0f);                         // outer([1,3,3,1])/64 * 4
  s += nwp[0] * noise[((unsigned)y << Hosh) + (unsigned)x] + ab[o];
  act[idx] = f2bf(flrelu(s));
}

// skip upsample: upfirdn2d(up=2, pad=(2,1), k=4x4) on f32 NCHW [8][3][H][H] -> [8][3][2H][2H]
__global__ void skip_up_k(const float* __restrict__ sin, float* __restrict__ sout, int Hsh) {
  unsigned H = 1u << Hsh, Ho = H << 1;
  int Hosh = Hsh + 1;
  unsigned idx = blockIdx.x * blockDim.x + threadIdx.x;
  unsigned total = 24u << (2 * Hosh);
  if (idx >= total) return;
  int x = (int)(idx & (Ho - 1u));
  int y = (int)((idx >> Hosh) & (Ho - 1u));
  unsigned rest = idx >> (2 * Hosh);
  unsigned c = rest % 3u;                      // constant divisors
  unsigned b = rest / 3u;
  const float BB[4] = {1.f, 3.f, 3.f, 1.f};
  float s = 0.f;
  for (int ky = 0; ky < 4; ++ky) {
    int u = y + ky - 2;
    if (u < 0 || (u & 1)) continue;
    u >>= 1;
    if (u >= (int)H) continue;
    for (int kx = 0; kx < 4; ++kx) {
      int v = x + kx - 2;
      if (v < 0 || (v & 1)) continue;
      v >>= 1;
      if (v >= (int)H) continue;
      s += BB[ky] * BB[kx] * sin[(((b * 3u + c) << Hsh) + (unsigned)u) * H + (unsigned)v];
    }
  }
  sout[idx] = s * (1.0f / 16.0f);
}

// to_rgb: 1x1 modulated conv (no demod) + bias (+ optional skip add) -> f32 NCHW
__global__ void to_rgb_k(const unsigned short* __restrict__ act, const float* __restrict__ styleR,
                         const float* __restrict__ w, const float* __restrict__ bias,
                         const float* __restrict__ skipAdd, float* __restrict__ out,
                         int HHsh, int Ish) {
  unsigned idx = blockIdx.x * blockDim.x + threadIdx.x;
  unsigned total = 8u << HHsh;
  if (idx >= total) return;
  unsigned HH = 1u << HHsh, I = 1u << Ish;
  unsigned pix = idx & (HH - 1u);
  unsigned b = idx >> HHsh;
  const unsigned short* ap = act + ((size_t)idx << Ish);
  const float* st = styleR + (b << Ish);
  float a0 = 0.f, a1 = 0.f, a2 = 0.f;
  for (unsigned i = 0; i < I; ++i) {
    float v = bf2f(ap[i]) * st[i];
    a0 += v * w[i];
    a1 += v * w[I + i];
    a2 += v * w[2 * I + i];
  }
  unsigned ob = ((b * 3u) << HHsh) + pix;
  float r0 = a0 + bias[0], r1 = a1 + bias[1], r2 = a2 + bias[2];
  if (skipAdd) { r0 += skipAdd[ob]; r1 += skipAdd[ob + HH]; r2 += skipAdd[ob + 2 * HH]; }
  out[ob] = r0; out[ob + HH] = r1; out[ob + 2 * HH] = r2;
}

// ============================ host orchestration ============================

static inline unsigned gdiv(long n, int b) { return (unsigned)((n + b - 1) / b); }
static inline int ilog2i(int v) { int s = 0; while ((1 << s) < v) ++s; return s; }

static void run_conv_layer(const float* latent,
                           const float* w, const float* mw, const float* mb,
                           const float* nwp, const float* ab, const float* noise,
                           const void* actIn, bool constSrc,
                           unsigned short* actOut, unsigned short* tmp,
                           float* styleS, float* wsq, float* demod, unsigned short* wpk,
                           unsigned short* xm, int H, int I, int O, bool up,
                           hipStream_t stream) {
  const int Ish = ilog2i(I), Osh = ilog2i(O), Hsh = ilog2i(H);
  float cscale = 1.0f / sqrtf((float)(I * 9));
  style_k<<<gdiv(8L * I, 256), 256, 0, stream>>>(latent, mw, mb, styleS, Ish, cscale);
  wsq_k<<<gdiv((long)O * I, 256), 256, 0, stream>>>(w, wsq, (unsigned)(O * I));
  demod_k<<<gdiv(8L * O, 256), 256, 0, stream>>>(styleS, wsq, demod, Ish, Osh);
  pack_k<<<gdiv((long)O * 9 * I, 256), 256, 0, stream>>>(w, wpk, Ish, up ? 1 : 0,
                                                         (unsigned)(O * 9 * I));
  long nmod = 8L * H * H * I;
  if (constSrc) modulate_k<true ><<<gdiv(nmod, 256), 256, 0, stream>>>(actIn, styleS, xm, 2 * Hsh, Ish);
  else          modulate_k<false><<<gdiv(nmod, 256), 256, 0, stream>>>(actIn, styleS, xm, 2 * Hsh, Ish);
  int Ho = up ? 2 * H + 1 : H;
  int otg_sh = ilog2i(O / 128);               // O in {128,256,512} -> {0,1,2}
  dim3 grid(gdiv(Ho, 32), (unsigned)Ho, 8u << otg_sh);
  if (up) {
    conv_wmma_k<true><<<grid, 256, 0, stream>>>(xm, wpk, demod, nullptr, nullptr, nullptr,
                                                tmp, H, I, O, otg_sh);
    long nb = 8L * (2 * H) * (2 * H) * O;
    blur_k<<<gdiv(nb, 256), 256, 0, stream>>>(tmp, noise, ab, nwp, actOut, Ho, Hsh + 1, Osh);
  } else {
    conv_wmma_k<false><<<grid, 256, 0, stream>>>(xm, wpk, demod, noise, ab, nwp,
                                                 actOut, H, I, O, otg_sh);
  }
}

extern "C" void kernel_launch(void* const* d_in, const int* in_sizes, int n_in,
                              void* d_out, int out_size, void* d_ws, size_t ws_size,
                              hipStream_t stream) {
  // ---- resolve input flattening order ----
  // insertion order: in_sizes[1] == 512 (z).  jax-pytree sorted keys: in_sizes[1] == 16 (noise 4x4).
  int I_label, I_z, I_mlp_w[8], I_mlp_b[8], I_const;
  int C_w[13], C_mw[13], C_mb[13], C_nw[13], C_ab[13];   // 0 = conv1, 1..12 = convs[]
  int R_w[7], R_mw[7], R_mb[7], R_bias[7];               // 0 = rgb1, 1..6 = rgbs[]
  int I_noise[13];
  bool sorted = (n_in > 1 && in_sizes[1] == 16);
  if (!sorted) {
    int k = 0;
    I_label = k++; I_z = k++;
    for (int i = 0; i < 8; ++i) I_mlp_w[i] = k++;
    for (int i = 0; i < 8; ++i) I_mlp_b[i] = k++;
    I_const = k++;
    C_w[0] = k; C_mw[0] = k + 1; C_mb[0] = k + 2; C_nw[0] = k + 3; C_ab[0] = k + 4; k += 5;
    R_w[0] = k; R_mw[0] = k + 1; R_mb[0] = k + 2; R_bias[0] = k + 3; k += 4;
    for (int l = 1; l <= 12; ++l) { C_w[l] = k; C_mw[l] = k + 1; C_mb[l] = k + 2; C_nw[l] = k + 3; C_ab[l] = k + 4; k += 5; }
    for (int r = 1; r <= 6; ++r) { R_w[r] = k; R_mw[r] = k + 1; R_mb[r] = k + 2; R_bias[r] = k + 3; k += 4; }
    for (int i = 0; i < 13; ++i) I_noise[i] = k++;
  } else {
    int k = 0;
    I_label = k++;
    for (int i = 0; i < 13; ++i) I_noise[i] = k++;
    I_const = k++;
    // conv dict sorted: ab, mb, mw, nw, w
    for (int l = 0; l <= 12; ++l) { C_ab[l] = k; C_mb[l] = k + 1; C_mw[l] = k + 2; C_nw[l] = k + 3; C_w[l] = k + 4; k += 5; }
    for (int i = 0; i < 8; ++i) I_mlp_b[i] = k++;
    for (int i = 0; i < 8; ++i) I_mlp_w[i] = k++;
    // rgb dict sorted: bias, mb, mw, w
    for (int r = 0; r <= 6; ++r) { R_bias[r] = k; R_mb[r] = k + 1; R_mw[r] = k + 2; R_w[r] = k + 3; k += 4; }
    I_z = k++;
  }
  auto F = [&](int i) { return (const float*)d_in[i]; };

  // ---- workspace layout (bump allocator, 256B aligned) ----
  char* wsb = (char*)d_ws;
  size_t off = 0;
  auto alloc = [&](size_t bytes) -> void* {
    void* p = wsb + off;
    off = (off + bytes + 255) & ~(size_t)255;
    return p;
  };
  float* latent = (float*)alloc(8 * 320 * 4);
  float* styleS = (float*)alloc(8 * 512 * 4);
  float* styleR = (float*)alloc(8 * 512 * 4);
  float* demod  = (float*)alloc(8 * 512 * 4);
  float* wsq    = (float*)alloc((size_t)512 * 512 * 4);
  unsigned short* wpk = (unsigned short*)alloc((size_t)512 * 9 * 512 * 2);
  const size_t actMax = (size_t)8 * 256 * 256 * 128;               // bf16 elems, max layer
  unsigned short* actA = (unsigned short*)alloc(actMax * 2);
  unsigned short* actB = (unsigned short*)alloc(actMax * 2);
  unsigned short* xm   = (unsigned short*)alloc(actMax * 2);
  unsigned short* tmp  = (unsigned short*)alloc((size_t)8 * 257 * 257 * 128 * 2);
  float* skipA = (float*)alloc((size_t)8 * 3 * 256 * 256 * 4);
  float* skipB = (float*)alloc((size_t)8 * 3 * 256 * 256 * 4);
  (void)ws_size; (void)out_size;

  // ---- mapping network -> latent [8,320] ----
  P8 W, Bp;
  for (int i = 0; i < 8; ++i) { W.p[i] = F(I_mlp_w[i]); Bp.p[i] = F(I_mlp_b[i]); }
  mapping_k<<<1, 512, 0, stream>>>(F(I_label), F(I_z), W, Bp, latent);

  // channels per resolution 4,8,16,32,64,128,256
  const int CH[7] = {512, 512, 512, 512, 512, 256, 128};

  // ---- conv1 @ 4x4 (input: const, f32 NCHW, batch-broadcast) ----
  run_conv_layer(latent, F(C_w[0]), F(C_mw[0]), F(C_mb[0]), F(C_nw[0]), F(C_ab[0]),
                 F(I_noise[0]), (const void*)F(I_const), /*constSrc=*/true,
                 actA, nullptr, styleS, wsq, demod, wpk, xm,
                 /*H=*/4, /*I=*/512, /*O=*/512, /*up=*/false, stream);

  // ---- rgb1 -> skipA @ 4x4 ----
  {
    float sc = 1.0f / sqrtf(512.0f);
    style_k<<<gdiv(8L * 512, 256), 256, 0, stream>>>(latent, F(R_mw[0]), F(R_mb[0]), styleR, 9, sc);
    to_rgb_k<<<gdiv(8L * 4 * 4, 128), 128, 0, stream>>>(actA, styleR, F(R_w[0]), F(R_bias[0]),
                                                        nullptr, skipA, 4, 9);
  }

  // ---- pyramid: 6 (up-conv, conv, skip-upsample, to_rgb) stages ----
  int ni = 1;
  for (int j = 0; j < 6; ++j) {
    int H = 4 << j;
    int ic = CH[j], oc = CH[j + 1];
    int l_up = 1 + 2 * j, l_cv = 2 + 2 * j;
    // up conv: actA (H, ic) -> tmp (2H+1, oc) -> blur -> actB (2H, oc)
    run_conv_layer(latent, F(C_w[l_up]), F(C_mw[l_up]), F(C_mb[l_up]), F(C_nw[l_up]), F(C_ab[l_up]),
                   F(I_noise[ni]), actA, false, actB, tmp,
                   styleS, wsq, demod, wpk, xm, H, ic, oc, /*up=*/true, stream);
    // conv: actB (2H, oc) -> actA (2H, oc)
    run_conv_layer(latent, F(C_w[l_cv]), F(C_mw[l_cv]), F(C_mb[l_cv]), F(C_nw[l_cv]), F(C_ab[l_cv]),
                   F(I_noise[ni + 1]), actB, false, actA, nullptr,
                   styleS, wsq, demod, wpk, xm, 2 * H, oc, oc, /*up=*/false, stream);
    ni += 2;
    // skip upsample: skipA (H) -> skipB (2H)
    skip_up_k<<<gdiv(8L * 3 * (2 * H) * (2 * H), 256), 256, 0, stream>>>(skipA, skipB, ilog2i(H));
    // to_rgb: actA (2H) + skipB -> skipA (or d_out on last stage)
    float sc = 1.0f / sqrtf((float)oc);
    style_k<<<gdiv(8L * oc, 256), 256, 0, stream>>>(latent, F(R_mw[j + 1]), F(R_mb[j + 1]),
                                                    styleR, ilog2i(oc), sc);
    float* dst = (j == 5) ? (float*)d_out : skipA;
    to_rgb_k<<<gdiv(8L * (2 * H) * (2 * H), 128), 128, 0, stream>>>(
        actA, styleR, F(R_w[j + 1]), F(R_bias[j + 1]), skipB, dst,
        ilog2i(2 * H) * 2, ilog2i(oc));
  }
}
